// DiagonalSSMLayer_62302795596560
// MI455X (gfx1250) — matrix operand order repaired and checked by reading, not compiled, not run on hardware
//
#include <hip/hip_runtime.h>
#include <hip/hip_bf16.h>
#include <math.h>

// ---------------------------------------------------------------------------
// DiagonalSSM layer for MI455X (gfx1250, wave32, WMMA + async global->LDS).
//   0) f32 -> bf16 pre-pass for x and weights
//   1) a = sigmoid(x@Wa^T + bias), b = x@B^T     (WMMA, LDS-staged A tile)
//   2) chunked parallel scan (f32), emits h as bf16
//   3) y = h@C^T + x@D^T                          (WMMA, LDS-staged A tile)
// A-tiles shared by all waves in a block are staged through LDS with
// double-buffered async copies (ASYNCcnt) overlapping the WMMA pipe.
// ---------------------------------------------------------------------------

#define D_MODEL 1024
#define STATE   256
#define BATCH   8
#define SEQ     4096
#define TOKENS  (BATCH * SEQ)          // 32768
#define CHUNK   256
#define NCHUNK  (SEQ / CHUNK)          // 16
#define APAD    40                     // LDS row stride (bf16): 32 + 8 pad

typedef __attribute__((ext_vector_type(16))) __bf16 v16bf;
typedef __attribute__((ext_vector_type(8)))  __bf16 v8bf;
typedef __attribute__((ext_vector_type(8)))  float  v8f;
typedef int v4i_ __attribute__((vector_size(16)));

__device__ __forceinline__ __bf16 f2bf(float f) {           // RNE, pre-pass and scan fixup
    union { float f; unsigned u; } in; in.f = f;
    unsigned r = in.u + 0x7FFFu + ((in.u >> 16) & 1u);
    union { unsigned short s; __bf16 b; } out;
    out.s = (unsigned short)(r >> 16);
    return out.b;
}

__device__ __forceinline__ v16bf cat16(v8bf lo, v8bf hi) {
    return __builtin_shufflevector(lo, hi, 0,1,2,3,4,5,6,7,8,9,10,11,12,13,14,15);
}

// A-operand (M=16 x K=32): lanes 0-15 hold K 0..15, lanes 16-31 K 16..31
__device__ __forceinline__ v16bf load_a16(const __bf16* row, int half) {
    v8bf lo = *(const v8bf*)(row + half * 16);
    v8bf hi = *(const v8bf*)(row + half * 16 + 8);
    return cat16(lo, hi);
}

// B-operand (K=32 x N=16): lane holds column n=lane&15, 16 contiguous K at half*16
__device__ __forceinline__ v16bf load_b16(const __bf16* __restrict__ p) {
    v8bf lo = *(const v8bf*)(p);
    v8bf hi = *(const v8bf*)(p + 8);
    return cat16(lo, hi);
}

__device__ __forceinline__ v8f bf_wmma(v16bf a, v16bf b, v8f c) {
    return __builtin_amdgcn_wmma_f32_16x16x32_bf16(
        false, a, false, b, (short)0, c, false, false);
}

// ---- async global -> LDS copy (16B per lane), with safe fallback ----------
#if __has_builtin(__builtin_amdgcn_global_load_async_to_lds_b128) && \
    __has_builtin(__builtin_amdgcn_s_wait_asynccnt)
#define HAVE_ASYNC_LDS 1
#else
#define HAVE_ASYNC_LDS 0
#endif

__device__ __forceinline__ void copy16_g2l(__bf16* lds_dst, const __bf16* gsrc) {
#if HAVE_ASYNC_LDS
    __builtin_amdgcn_global_load_async_to_lds_b128(
        (__attribute__((address_space(1))) v4i_*)gsrc,
        (__attribute__((address_space(3))) v4i_*)lds_dst, 0, 0);
#else
    *(v8bf*)lds_dst = *(const v8bf*)gsrc;
#endif
}

__device__ __forceinline__ void copy_wait() {
#if HAVE_ASYNC_LDS
    __builtin_amdgcn_s_wait_asynccnt(0);
#endif
}

// ---------------------------------------------------------------------------
// Pre-pass: f32 -> bf16, 8 elements/thread.
// ---------------------------------------------------------------------------
__global__ __launch_bounds__(256) void cvt_bf16_kernel(
    const float* __restrict__ src, __bf16* __restrict__ dst)
{
    size_t i = ((size_t)blockIdx.x * 256 + threadIdx.x) * 8;
    float4 u = *(const float4*)(src + i);
    float4 v = *(const float4*)(src + i + 4);
    v8bf r;
    r[0] = f2bf(u.x); r[1] = f2bf(u.y); r[2] = f2bf(u.z); r[3] = f2bf(u.w);
    r[4] = f2bf(v.x); r[5] = f2bf(v.y); r[6] = f2bf(v.z); r[7] = f2bf(v.w);
    *(v8bf*)(dst + i) = r;
}

// ---------------------------------------------------------------------------
// Kernel 1: a/b projections. Block = 32 tokens x 128 states, 8 waves; the
// 32x32 A tile (shared by ALL waves) is double-buffer staged in LDS.
// grid = (TOKENS/32)*(STATE/128) = 2048 blocks.
// ---------------------------------------------------------------------------
__global__ __launch_bounds__(256) void ab_gemm_kernel(
    const __bf16* __restrict__ xbf, const __bf16* __restrict__ wabf,
    const float* __restrict__ Wa_b, const __bf16* __restrict__ bwbf,
    float* __restrict__ a_out, float* __restrict__ b_out)
{
    __shared__ __bf16 sA[2][32 * APAD];          // 5 KB

    const int tid  = threadIdx.x;
    const int lane = tid & 31;
    const int wave = tid >> 5;                   // nTile within block
    const int half = lane >> 4;
    const int l16  = lane & 15;

    const int tokBlk = blockIdx.x >> 1;          // 0..1023  (32 tokens)
    const int nBlk   = blockIdx.x & 1;           // 0..1     (128 states)
    const int tokBase = tokBlk * 32;
    const int col = nBlk * 128 + wave * 16 + l16;

    const __bf16* srcA = xbf + (size_t)tokBase * D_MODEL;
    const __bf16* wa = wabf + (size_t)col * D_MODEL;
    const __bf16* bw = bwbf + (size_t)col * D_MODEL;

    const int srow = tid >> 2;                   // staging: 128 threads, 16B each
    const int sseg = tid & 3;

    if (tid < 128)
        copy16_g2l(&sA[0][srow * APAD + sseg * 8],
                   srcA + (size_t)srow * D_MODEL + sseg * 8);

    v8f accA0 = {}; v8f accA1 = {}; v8f accB0 = {}; v8f accB1 = {};
    const int NK = D_MODEL / 32;
    for (int i = 0; i < NK; ++i) {
        const int kt = i * 32;
        copy_wait();
        __syncthreads();                          // sA[i&1] ready everywhere
        if (i + 1 < NK && tid < 128)
            copy16_g2l(&sA[(i + 1) & 1][srow * APAD + sseg * 8],
                       srcA + (size_t)srow * D_MODEL + kt + 32 + sseg * 8);

        const __bf16* ab = &sA[i & 1][0];
        v16bf A0  = load_a16(ab + (l16)      * APAD, half);
        v16bf A1  = load_a16(ab + (16 + l16) * APAD, half);
        v16bf Bwa = load_b16(wa + kt + half * 16);
        v16bf Bb  = load_b16(bw + kt + half * 16);
        accA0 = bf_wmma(A0, Bwa, accA0);
        accA1 = bf_wmma(A1, Bwa, accA1);
        accB0 = bf_wmma(A0, Bb,  accB0);
        accB1 = bf_wmma(A1, Bb,  accB1);
    }

    const float bias = Wa_b[col];
#pragma unroll
    for (int r = 0; r < 8; ++r) {
        const int row0 = tokBase + half * 8 + r;
        const int row1 = row0 + 16;
        float a0 = accA0[r] + bias;
        float a1 = accA1[r] + bias;
        a_out[(size_t)row0 * STATE + col] =
            __builtin_amdgcn_rcpf(1.0f + __expf(-a0));
        a_out[(size_t)row1 * STATE + col] =
            __builtin_amdgcn_rcpf(1.0f + __expf(-a1));
        b_out[(size_t)row0 * STATE + col] = accB0[r];
        b_out[(size_t)row1 * STATE + col] = accB1[r];
    }
}

// ---------------------------------------------------------------------------
// Kernel 2: per-chunk local scan (in place, f32).
// ---------------------------------------------------------------------------
__global__ __launch_bounds__(256) void scan_chunks_kernel(
    float* __restrict__ a_ws, float* __restrict__ b_ws,
    float* __restrict__ aggA, float* __restrict__ aggB)
{
    const int batch = blockIdx.x >> 4;
    const int chunk = blockIdx.x & 15;
    const int n = threadIdx.x;
    size_t base = ((size_t)batch * SEQ + (size_t)chunk * CHUNK) * STATE + n;

    float cumA = 1.0f, h = 0.0f;
    for (int t = 0; t < CHUNK; ++t) {
        size_t idx = base + (size_t)t * STATE;
        float a = a_ws[idx];
        float b = b_ws[idx];
        h = fmaf(a, h, b);
        cumA *= a;
        a_ws[idx] = cumA;
        b_ws[idx] = h;
    }
    size_t agg = ((size_t)batch * NCHUNK + chunk) * STATE + n;
    aggA[agg] = cumA;
    aggB[agg] = h;
}

// ---------------------------------------------------------------------------
// Kernel 3: scan chunk aggregates -> exclusive carry-in per chunk (in aggB).
// ---------------------------------------------------------------------------
__global__ __launch_bounds__(256) void scan_agg_kernel(
    const float* __restrict__ aggA, float* __restrict__ aggB)
{
    const int batch = blockIdx.x;
    const int n = threadIdx.x;
    float carry = 0.0f;
    for (int c = 0; c < NCHUNK; ++c) {
        size_t i = ((size_t)batch * NCHUNK + c) * STATE + n;
        float nc = fmaf(aggA[i], carry, aggB[i]);
        aggB[i] = carry;
        carry = nc;
    }
}

// ---------------------------------------------------------------------------
// Kernel 4: fixup h = h_local + cumA*carry_in, emit bf16 h for the y-GEMM.
// ---------------------------------------------------------------------------
__global__ __launch_bounds__(256) void scan_fix_kernel(
    const float* __restrict__ a_ws, const float* __restrict__ b_ws,
    const float* __restrict__ aggB, __bf16* __restrict__ hbf)
{
    size_t i = ((size_t)blockIdx.x * 256 + threadIdx.x) * 8; // ((b*SEQ+t)*STATE+n)
    int n = (int)(i & (STATE - 1));
    size_t bt = i >> 8;
    int t = (int)(bt & (SEQ - 1));
    int b = (int)(bt >> 12);
    int chunk = t >> 8;
    const float* cw = aggB + ((size_t)b * NCHUNK + chunk) * STATE + n;

    v8bf r;
#pragma unroll
    for (int j = 0; j < 8; j += 4) {
        float4 a = *(const float4*)(a_ws + i + j);
        float4 h = *(const float4*)(b_ws + i + j);
        float4 c = *(const float4*)(cw + j);
        r[j]     = f2bf(fmaf(a.x, c.x, h.x));
        r[j + 1] = f2bf(fmaf(a.y, c.y, h.y));
        r[j + 2] = f2bf(fmaf(a.z, c.z, h.z));
        r[j + 3] = f2bf(fmaf(a.w, c.w, h.w));
    }
    *(v8bf*)(hbf + i) = r;
}

// ---------------------------------------------------------------------------
// Kernel 5: y = h@C^T + x@D^T. Block = 64 tokens x 128 d-cols, 8 waves as
// 2 (tok) x 4 (col); the 64x32 A tile is double-buffer staged in LDS and
// shared by the 4 column-waves. grid = (TOKENS/64)*(D_MODEL/128) = 4096.
// ---------------------------------------------------------------------------
__global__ __launch_bounds__(256) void y_gemm_kernel(
    const __bf16* __restrict__ hbf, const __bf16* __restrict__ xbf,
    const __bf16* __restrict__ cbf, const __bf16* __restrict__ dbf,
    float* __restrict__ y)
{
    __shared__ __bf16 sA[2][64 * APAD];          // 10 KB

    const int tid  = threadIdx.x;
    const int lane = tid & 31;
    const int wave = tid >> 5;
    const int half = lane >> 4;
    const int l16  = lane & 15;
    const int tokGrp = wave >> 2;                // 0..1
    const int colGrp = wave & 3;                 // 0..3

    const int tokBlk = blockIdx.x >> 3;          // 0..511  (64 tokens)
    const int colBlk = blockIdx.x & 7;           // 0..7    (128 cols)
    const int tokBase = tokBlk * 64;
    const int colBase = colBlk * 128 + colGrp * 32;

    const __bf16* c0 = cbf + (size_t)(colBase + l16)      * STATE;
    const __bf16* c1 = cbf + (size_t)(colBase + 16 + l16) * STATE;
    const __bf16* d0 = dbf + (size_t)(colBase + l16)      * D_MODEL;
    const __bf16* d1 = dbf + (size_t)(colBase + 16 + l16) * D_MODEL;

    const int srow = tid >> 2;                   // staging: 256 threads, 16B each
    const int sseg = tid & 3;
    const int arow0 = (tokGrp * 32 + l16) * APAD;
    const int arow1 = (tokGrp * 32 + 16 + l16) * APAD;

    v8f acc00 = {}; v8f acc01 = {}; v8f acc10 = {}; v8f acc11 = {};

    // ---- phase 1: h @ C^T (K = STATE) ----
    {
        const __bf16* srcA = hbf + (size_t)tokBase * STATE;
        copy16_g2l(&sA[0][srow * APAD + sseg * 8],
                   srcA + (size_t)srow * STATE + sseg * 8);
        const int NK = STATE / 32;
        for (int i = 0; i < NK; ++i) {
            const int kt = i * 32;
            copy_wait();
            __syncthreads();
            if (i + 1 < NK)
                copy16_g2l(&sA[(i + 1) & 1][srow * APAD + sseg * 8],
                           srcA + (size_t)srow * STATE + kt + 32 + sseg * 8);
            const __bf16* ab = &sA[i & 1][0];
            v16bf A0 = load_a16(ab + arow0, half);
            v16bf A1 = load_a16(ab + arow1, half);
            v16bf B0 = load_b16(c0 + kt + half * 16);
            v16bf B1 = load_b16(c1 + kt + half * 16);
            acc00 = bf_wmma(A0, B0, acc00);
            acc01 = bf_wmma(A0, B1, acc01);
            acc10 = bf_wmma(A1, B0, acc10);
            acc11 = bf_wmma(A1, B1, acc11);
        }
    }
    __syncthreads();

    // ---- phase 2: x @ D^T (K = D_MODEL) ----
    {
        const __bf16* srcA = xbf + (size_t)tokBase * D_MODEL;
        copy16_g2l(&sA[0][srow * APAD + sseg * 8],
                   srcA + (size_t)srow * D_MODEL + sseg * 8);
        const int NK = D_MODEL / 32;
        for (int i = 0; i < NK; ++i) {
            const int kt = i * 32;
            copy_wait();
            __syncthreads();
            if (i + 1 < NK)
                copy16_g2l(&sA[(i + 1) & 1][srow * APAD + sseg * 8],
                           srcA + (size_t)srow * D_MODEL + kt + 32 + sseg * 8);
            const __bf16* ab = &sA[i & 1][0];
            v16bf A0 = load_a16(ab + arow0, half);
            v16bf A1 = load_a16(ab + arow1, half);
            v16bf B0 = load_b16(d0 + kt + half * 16);
            v16bf B1 = load_b16(d1 + kt + half * 16);
            acc00 = bf_wmma(A0, B0, acc00);
            acc01 = bf_wmma(A0, B1, acc01);
            acc10 = bf_wmma(A1, B0, acc10);
            acc11 = bf_wmma(A1, B1, acc11);
        }
    }

    const int col0 = colBase + l16;
#pragma unroll
    for (int r = 0; r < 8; ++r) {
        const int row0 = tokBase + tokGrp * 32 + half * 8 + r;
        const int row1 = row0 + 16;
        y[(size_t)row0 * D_MODEL + col0]      = acc00[r];
        y[(size_t)row0 * D_MODEL + col0 + 16] = acc01[r];
        y[(size_t)row1 * D_MODEL + col0]      = acc10[r];
        y[(size_t)row1 * D_MODEL + col0 + 16] = acc11[r];
    }
}

// ---------------------------------------------------------------------------
extern "C" void kernel_launch(void* const* d_in, const int* in_sizes, int n_in,
                              void* d_out, int out_size, void* d_ws, size_t ws_size,
                              hipStream_t stream) {
    const float* x    = (const float*)d_in[0];
    const float* Wa_w = (const float*)d_in[1];
    const float* Wa_b = (const float*)d_in[2];
    const float* B_w  = (const float*)d_in[3];
    const float* C_w  = (const float*)d_in[4];
    const float* D_w  = (const float*)d_in[5];
    float* y = (float*)d_out;

    char* p = (char*)d_ws;
    float* a_ws = (float*)p;  p += (size_t)TOKENS * STATE * 4;      // 32 MB
    float* b_ws = (float*)p;  p += (size_t)TOKENS * STATE * 4;      // 32 MB
    float* aggA = (float*)p;  p += (size_t)BATCH * NCHUNK * STATE * 4;
    float* aggB = (float*)p;  p += (size_t)BATCH * NCHUNK * STATE * 4;
    __bf16* xbf  = (__bf16*)p; p += (size_t)TOKENS * D_MODEL * 2;   // 64 MB
    __bf16* hbf  = (__bf16*)p; p += (size_t)TOKENS * STATE * 2;     // 16 MB
    __bf16* wabf = (__bf16*)p; p += (size_t)STATE * D_MODEL * 2;
    __bf16* bwbf = (__bf16*)p; p += (size_t)STATE * D_MODEL * 2;
    __bf16* cbf  = (__bf16*)p; p += (size_t)D_MODEL * STATE * 2;
    __bf16* dbf  = (__bf16*)p; p += (size_t)D_MODEL * D_MODEL * 2;

    // 0) bf16 pre-passes
    cvt_bf16_kernel<<<(TOKENS * D_MODEL) / 2048, 256, 0, stream>>>(x,    xbf);
    cvt_bf16_kernel<<<(STATE * D_MODEL)  / 2048, 256, 0, stream>>>(Wa_w, wabf);
    cvt_bf16_kernel<<<(STATE * D_MODEL)  / 2048, 256, 0, stream>>>(B_w,  bwbf);
    cvt_bf16_kernel<<<(D_MODEL * STATE)  / 2048, 256, 0, stream>>>(C_w,  cbf);
    cvt_bf16_kernel<<<(D_MODEL * D_MODEL)/ 2048, 256, 0, stream>>>(D_w,  dbf);

    // 1) a/b projections (LDS-staged A)
    ab_gemm_kernel<<<(TOKENS / 32) * (STATE / 128), 256, 0, stream>>>(
        xbf, wabf, Wa_b, bwbf, a_ws, b_ws);

    // 2) scan
    scan_chunks_kernel<<<BATCH * NCHUNK, 256, 0, stream>>>(a_ws, b_ws, aggA, aggB);
    scan_agg_kernel   <<<BATCH, 256, 0, stream>>>(aggA, aggB);
    scan_fix_kernel   <<<(TOKENS * STATE) / 2048, 256, 0, stream>>>(a_ws, b_ws, aggB, hbf);

    // 3) y GEMM (LDS-staged A)
    y_gemm_kernel<<<(TOKENS / 64) * (D_MODEL / 128), 256, 0, stream>>>(
        hbf, xbf, cbf, dbf, y);
}
